// GroupedQueryAttention_85701777424534
// MI455X (gfx1250) — matrix operand order repaired and checked by reading, not compiled, not run on hardware
//
#include <hip/hip_runtime.h>
#include <hip/hip_bf16.h>
#include <stdint.h>
#include <stddef.h>

// ---------------- problem constants ----------------
#define T_LEN    4096
#define DMODEL   1024
#define NUMH     16
#define NUMKV    4
#define HD       64
#define KVDIM    (NUMKV * HD)          // 256
#define QKVN     (DMODEL + 2 * KVDIM)  // 1536
#define KSTEP    32
#define NEG_INF  (-1.7014118e38f)      // finfo(f32).min / 2

// ---------------- WMMA types (gfx1250, wave32) ----------------
typedef __attribute__((ext_vector_type(16))) __bf16 bf16x16;
typedef __attribute__((ext_vector_type(8)))  float  f32x8;
typedef __attribute__((ext_vector_type(4)))  unsigned int u32x4;

union Frag { bf16x16 bf; u32x4 q[2]; };

__device__ __forceinline__ u32x4 ld16(const __bf16* p) {
  return *(const u32x4*)p;
}

__device__ __forceinline__ f32x8 wmma_bf16(const Frag& a, const Frag& b, f32x8 c) {
  // D = A(16x32 bf16) x B(32x16 bf16) + C(16x16 f32)
  return __builtin_amdgcn_wmma_f32_16x16x32_bf16(
      /*neg_a=*/false, a.bf, /*neg_b=*/false, b.bf,
      /*c_mod=*/(short)0, c, /*reuse_a=*/false, /*reuse_b=*/false);
}

// gfx1250 async copy: per-lane 16B global -> LDS, tracked by ASYNCcnt.
__device__ __forceinline__ void async_ld_b128(const void* gptr, void* lptr) {
  unsigned long long ga = (unsigned long long)gptr;
  unsigned int       la = (unsigned int)(uintptr_t)lptr;  // generic low 32 = LDS offset
  asm volatile("global_load_async_to_lds_b128 %0, %1, off"
               :: "v"(la), "v"(ga) : "memory");
}
__device__ __forceinline__ void wait_async0() {
  asm volatile("s_wait_asynccnt 0x0" ::: "memory");
}

// ---------------- conversion kernels ----------------
__global__ __launch_bounds__(256) void f32_to_bf16_kernel(
    const float* __restrict__ in, __bf16* __restrict__ out, int n) {
  int i = blockIdx.x * blockDim.x + threadIdx.x;
  if (i < n) out[i] = (__bf16)in[i];
}

// out[n][k] = in[k][n]   (in: [K][N] f32 row-major -> out: [N][K] bf16)
__global__ __launch_bounds__(256) void transpose_to_bf16_kernel(
    const float* __restrict__ in, __bf16* __restrict__ out, int K, int N) {
  int i = blockIdx.x * blockDim.x + threadIdx.x;
  if (i < K * N) {
    int n = i / K;
    int k = i - n * K;
    out[i] = (__bf16)in[(size_t)k * N + n];
  }
}

// ---------------- RoPE + split kernel ----------------
// qkv_f32: [T][1536].  Writes:
//   qb [NUMH][T][64]  (RoPE applied, pre-scaled by 1/sqrt(64))
//   kb [NUMKV][T][64] (RoPE applied)
//   vT [NUMKV][64][T] (transposed for PV-stage B fragments)
__global__ __launch_bounds__(256) void rope_split_kernel(
    const float* __restrict__ qkv, const float* __restrict__ sinp,
    const float* __restrict__ cosp, __bf16* __restrict__ qb,
    __bf16* __restrict__ kb, __bf16* __restrict__ vT) {
  int i = blockIdx.x * blockDim.x + threadIdx.x;  // T * 24 * 64
  if (i >= T_LEN * 24 * 64) return;
  int d = i & 63;
  int h = (i >> 6) % 24;
  int t = i / (64 * 24);
  const float* row = qkv + (size_t)t * QKVN;
  if (h < NUMH) {                         // Q heads
    float v = row[h * HD + d];
    float other = (d < 32) ? -row[h * HD + d + 32] : row[h * HD + d - 32];
    float s = sinp[t * HD + d], c = cosp[t * HD + d];
    qb[((size_t)h * T_LEN + t) * HD + d] = (__bf16)((v * c + other * s) * 0.125f);
  } else if (h < NUMH + NUMKV) {          // K heads
    int kh = h - NUMH;
    float v = row[DMODEL + kh * HD + d];
    float other = (d < 32) ? -row[DMODEL + kh * HD + d + 32]
                           :  row[DMODEL + kh * HD + d - 32];
    float s = sinp[t * HD + d], c = cosp[t * HD + d];
    kb[((size_t)kh * T_LEN + t) * HD + d] = (__bf16)(v * c + other * s);
  } else {                                // V heads, transposed
    int vh = h - NUMH - NUMKV;
    vT[((size_t)vh * HD + d) * T_LEN + t] = (__bf16)row[DMODEL + KVDIM + vh * HD + d];
  }
}

// ---------------- WMMA GEMM with async-LDS double buffering ----------------
//   C[M][N] = A[M][K] * Bt[N][K]^T
// Block = 128 threads (4 waves); block tile 64(M) x 64(N); wave w owns rows w*16..w*16+15.
// Per K-step (32): A slab 64x32 bf16 (4KB) + B slab 64x32 (4KB) staged to LDS via
// global_load_async_to_lds_b128 (ASYNCcnt), double buffered.
__global__ __launch_bounds__(128) void gemm_bf16_kernel(
    const __bf16* __restrict__ A, const __bf16* __restrict__ Bt,
    float* __restrict__ C, int M, int N, int K) {
  __shared__ __bf16 Alds[2][64][KSTEP];
  __shared__ __bf16 Blds[2][64][KSTEP];

  const int tid  = threadIdx.x;
  const int wave = tid >> 5;
  const int lane = tid & 31;
  const int col  = lane & 15;
  const int hi   = lane >> 4;
  const int m0   = blockIdx.y * 64;
  const int n0   = blockIdx.x * 64;

  // Each thread stages two 16B chunks of A and two of B per K-step.
  // chunk c in [0,256): row = c>>2, k-offset = (c&3)*8 elements.
  const int ca = tid * 2, cb = tid * 2 + 1;
  const int ra = ca >> 2, ka = (ca & 3) * 8;
  const int rb = cb >> 2, kb2 = (cb & 3) * 8;

  auto stage = [&](int buf, int k0) {
    async_ld_b128(A  + (size_t)(m0 + ra) * K + k0 + ka,  &Alds[buf][ra][ka]);
    async_ld_b128(A  + (size_t)(m0 + rb) * K + k0 + kb2, &Alds[buf][rb][kb2]);
    async_ld_b128(Bt + (size_t)(n0 + ra) * K + k0 + ka,  &Blds[buf][ra][ka]);
    async_ld_b128(Bt + (size_t)(n0 + rb) * K + k0 + kb2, &Blds[buf][rb][kb2]);
  };

  f32x8 acc[4] = {};
  const int niter = K / KSTEP;
  stage(0, 0);
#pragma unroll 2                    // make `buf` a compile-time constant per copy
  for (int it = 0; it < niter; ++it) {
    const int buf = it & 1;
    wait_async0();        // this wave's copies into buf are done
    __syncthreads();      // => every wave's copies into buf are done
    if (it + 1 < niter) stage(1 - buf, (it + 1) * KSTEP);

    // ---- hoist ALL fragment loads so one s_wait_dscnt covers the WMMA quad ----
    Frag a, b[4];
    const __bf16* arow = &Alds[buf][wave * 16 + col][0];
    a.q[0] = ld16(arow + hi * 8);        // A 16x32: K-groups {hi*8, 16+hi*8}
    a.q[1] = ld16(arow + 16 + hi * 8);
#pragma unroll
    for (int nb = 0; nb < 4; ++nb) {     // B 32x16: K-groups {hi*8, 16+hi*8}
      const __bf16* brow = &Blds[buf][nb * 16 + col][0];
      b[nb].q[0] = ld16(brow + hi * 8);
      b[nb].q[1] = ld16(brow + 16 + hi * 8);
    }
#pragma unroll
    for (int nb = 0; nb < 4; ++nb) acc[nb] = wmma_bf16(a, b[nb], acc[nb]);

    __syncthreads();      // all reads of buf done before it is overwritten
  }
#pragma unroll
  for (int nb = 0; nb < 4; ++nb)
#pragma unroll
    for (int r = 0; r < 8; ++r)   // C layout: row = r + 8*hi, col = n0+nb*16+(lane&15)
      C[(size_t)(m0 + wave * 16 + r + 8 * hi) * N + n0 + nb * 16 + col] = acc[nb][r];
}

// ---------------- flash attention (one wave per head x 16-query tile) ----------------
__global__ __launch_bounds__(32) void attn_kernel(
    const __bf16* __restrict__ q,   // [NUMH][T][64]  (pre-scaled)
    const __bf16* __restrict__ k,   // [NUMKV][T][64]
    const __bf16* __restrict__ vT,  // [NUMKV][64][T]
    const int* __restrict__ doc,    // [T]
    __bf16* __restrict__ attn)      // [T][DMODEL]
{
  __shared__ __bf16 plds[16][40];   // P transpose staging; 80B row stride (16B aligned)
  const int head = blockIdx.x;
  const int t0   = blockIdx.y * 16;
  const int kvh  = head / (NUMH / NUMKV);
  const int lane = threadIdx.x;
  const int col  = lane & 15;
  const int hi   = lane >> 4;

  const __bf16* qh = q  + ((size_t)head * T_LEN + t0) * HD;
  const __bf16* kh = k  + (size_t)kvh * T_LEN * HD;
  const __bf16* vh = vT + (size_t)kvh * HD * T_LEN;

  // Q A-fragments for d-chunks 0 (d=0..31) and 1 (d=32..63)
  Frag qa[2];
#pragma unroll
  for (int dc = 0; dc < 2; ++dc) {
    const __bf16* row = qh + (size_t)col * HD + dc * 32;
    qa[dc].q[0] = ld16(row + hi * 16);
    qa[dc].q[1] = ld16(row + hi * 16 + 8);
  }

  int doc_row[8];
#pragma unroll
  for (int r = 0; r < 8; ++r) doc_row[r] = doc[t0 + r + 8 * hi];

  f32x8 o[4] = {};
  float m_i[8], l_i[8];
#pragma unroll
  for (int r = 0; r < 8; ++r) { m_i[r] = NEG_INF; l_i[r] = 0.f; }

  for (int s0 = 0; s0 < t0 + 16; s0 += 32) {
    // ---- issue ALL global loads for this tile first (K frags, V frags, doc) ----
    Frag kf[2][2];            // [score-col-block][d-chunk]
#pragma unroll
    for (int dc = 0; dc < 2; ++dc) {
      const __bf16* kp0 = kh + (size_t)(s0 + col) * HD + dc * 32 + hi * 16;
      const __bf16* kp1 = kh + (size_t)(s0 + 16 + col) * HD + dc * 32 + hi * 16;
      kf[0][dc].q[0] = ld16(kp0); kf[0][dc].q[1] = ld16(kp0 + 8);
      kf[1][dc].q[0] = ld16(kp1); kf[1][dc].q[1] = ld16(kp1 + 8);
    }
    Frag vf[4];               // V B-fragments (used only after softmax -> latency hidden)
#pragma unroll
    for (int db = 0; db < 4; ++db) {
      const __bf16* vp = vh + (size_t)(db * 16 + col) * T_LEN + s0 + hi * 16;
      vf[db].q[0] = ld16(vp); vf[db].q[1] = ld16(vp + 8);
    }
    const int s_c0 = s0 + col, s_c1 = s0 + 16 + col;
    const int d_c0 = doc[s_c0], d_c1 = doc[s_c1];

    // ---- S = Q K^T over 32 keys (two 16-col score tiles) ----
    f32x8 sc0 = {}, sc1 = {};
#pragma unroll
    for (int dc = 0; dc < 2; ++dc) {
      sc0 = wmma_bf16(qa[dc], kf[0][dc], sc0);
      sc1 = wmma_bf16(qa[dc], kf[1][dc], sc1);
    }

    // ---- causal + document mask, online softmax ----
    float mnew[8], alpha[8];
#pragma unroll
    for (int r = 0; r < 8; ++r) {
      const int trow = t0 + r + 8 * hi;
      float v0 = ((s_c0 <= trow) && (d_c0 == doc_row[r])) ? sc0[r] : NEG_INF;
      float v1 = ((s_c1 <= trow) && (d_c1 == doc_row[r])) ? sc1[r] : NEG_INF;
      sc0[r] = v0; sc1[r] = v1;
      float mx = fmaxf(v0, v1);
      for (int off = 8; off; off >>= 1) mx = fmaxf(mx, __shfl_xor(mx, off, 32));
      mnew[r]  = fmaxf(m_i[r], mx);
      alpha[r] = __expf(m_i[r] - mnew[r]);
      m_i[r]   = mnew[r];
    }
#pragma unroll
    for (int r = 0; r < 8; ++r) {
      float p0 = __expf(sc0[r] - mnew[r]);
      float p1 = __expf(sc1[r] - mnew[r]);
      float rs = p0 + p1;
      for (int off = 8; off; off >>= 1) rs += __shfl_xor(rs, off, 32);
      l_i[r] = l_i[r] * alpha[r] + rs;
      plds[r + 8 * hi][col]      = (__bf16)p0;   // C-layout -> LDS (row-major)
      plds[r + 8 * hi][16 + col] = (__bf16)p1;
#pragma unroll
      for (int db = 0; db < 4; ++db) o[db][r] *= alpha[r];
    }
    // Same-wave DS ops are in-order; wait for store data before re-reading.
    asm volatile("s_wait_dscnt 0" ::: "memory");

    // ---- reload P as A-fragment (16x32), then O += P @ V ----
    Frag pa;
    pa.q[0] = ld16(&plds[col][hi * 16]);
    pa.q[1] = ld16(&plds[col][hi * 16 + 8]);
#pragma unroll
    for (int db = 0; db < 4; ++db) o[db] = wmma_bf16(pa, vf[db], o[db]);
  }

  // ---- normalize and write [T][DMODEL] bf16 ----
  float rl[8];
#pragma unroll
  for (int r = 0; r < 8; ++r) rl[r] = 1.0f / l_i[r];
#pragma unroll
  for (int db = 0; db < 4; ++db)
#pragma unroll
    for (int r = 0; r < 8; ++r) {
      int trow = t0 + r + 8 * hi;
      attn[(size_t)trow * DMODEL + head * HD + db * 16 + col] =
          (__bf16)(o[db][r] * rl[r]);
    }
}

// ---------------- launcher ----------------
extern "C" void kernel_launch(void* const* d_in, const int* in_sizes, int n_in,
                              void* d_out, int out_size, void* d_ws, size_t ws_size,
                              hipStream_t stream) {
  (void)in_sizes; (void)n_in; (void)out_size; (void)ws_size;
  const float* x    = (const float*)d_in[0];
  const float* sinp = (const float*)d_in[1];
  const float* cosp = (const float*)d_in[2];
  const float* Wqkv = (const float*)d_in[3];   // [1024][1536]
  const float* Wout = (const float*)d_in[4];   // [1024][1024]
  const int*   doc  = (const int*)d_in[5];     // [T]
  float* out = (float*)d_out;                  // [T][1024] f32

  char* ws = (char*)d_ws;
  size_t off = 0;
  auto take = [&](size_t bytes) { size_t o = off; off += (bytes + 255) & ~(size_t)255; return o; };
  __bf16* xb     = (__bf16*)(ws + take((size_t)T_LEN * DMODEL * 2));
  __bf16* wqkvT  = (__bf16*)(ws + take((size_t)QKVN * DMODEL * 2));
  __bf16* woutT  = (__bf16*)(ws + take((size_t)DMODEL * DMODEL * 2));
  float*  qkvf   = (float*) (ws + take((size_t)T_LEN * QKVN * 4));
  __bf16* qb     = (__bf16*)(ws + take((size_t)NUMH * T_LEN * HD * 2));
  __bf16* kb     = (__bf16*)(ws + take((size_t)NUMKV * T_LEN * HD * 2));
  __bf16* vT     = (__bf16*)(ws + take((size_t)NUMKV * HD * T_LEN * 2));
  __bf16* attnb  = (__bf16*)(ws + take((size_t)T_LEN * DMODEL * 2));

  // 1) precision conversion / weight transposition
  {
    int n = T_LEN * DMODEL;
    f32_to_bf16_kernel<<<(n + 255) / 256, 256, 0, stream>>>(x, xb, n);
  }
  {
    int n = DMODEL * QKVN;
    transpose_to_bf16_kernel<<<(n + 255) / 256, 256, 0, stream>>>(Wqkv, wqkvT, DMODEL, QKVN);
  }
  {
    int n = DMODEL * DMODEL;
    transpose_to_bf16_kernel<<<(n + 255) / 256, 256, 0, stream>>>(Wout, woutT, DMODEL, DMODEL);
  }

  // 2) QKV projection (WMMA + async-LDS): qkv_f32[T][1536] = xb @ Wqkv
  gemm_bf16_kernel<<<dim3(QKVN / 64, T_LEN / 64), 128, 0, stream>>>(
      xb, wqkvT, qkvf, T_LEN, QKVN, DMODEL);

  // 3) RoPE + head split (+ V transpose, + score scale folded into Q)
  {
    int n = T_LEN * 24 * 64;
    rope_split_kernel<<<(n + 255) / 256, 256, 0, stream>>>(qkvf, sinp, cosp, qb, kb, vT);
  }

  // 4) flash attention with causal + doc mask (WMMA QK^T and PV)
  attn_kernel<<<dim3(NUMH, T_LEN / 16), 32, 0, stream>>>(qb, kb, vT, doc, attnb);

  // 5) output projection (WMMA + async-LDS): out[T][1024] = attnb @ Wout
  gemm_bf16_kernel<<<dim3(DMODEL / 64, T_LEN / 64), 128, 0, stream>>>(
      attnb, woutT, out, T_LEN, DMODEL, DMODEL);
}
